// Attention_6184752906630
// MI455X (gfx1250) — matrix operand order repaired and checked by reading, not compiled, not run on hardware
//
#include <hip/hip_runtime.h>
#include <hip/hip_bf16.h>

// MHA for B=8, N=1024, D=768, H=12, DH=64 on gfx1250 (MI455X).
// Compute-bound (~64 GFLOP, ~60MB working set << 192MB L2) -> all matmuls via
// v_wmma_f32_16x16x32_f16 (f16 operands, f32 accumulate).
// GEMMs use 32x64 tiles per wave: 8 WMMAs per 12 b128 loads, A frags reuse
// each B (weight) fragment to cut L2 fragment traffic.

typedef __attribute__((ext_vector_type(4)))  float    v4f;
typedef __attribute__((ext_vector_type(8)))  _Float16 v8h;
typedef __attribute__((ext_vector_type(16))) _Float16 v16h;
typedef __attribute__((ext_vector_type(8)))  float    v8f;

#define AH  12
#define AD  768
#define ADH 64
#define AB  8
#define AN  1024
#define AM  (AB * AN)          // 8192 rows total

static __device__ __forceinline__ v16h cat8(v8h lo, v8h hi) {
  return __builtin_shufflevector(lo, hi, 0,1,2,3,4,5,6,7,8,9,10,11,12,13,14,15);
}

static __device__ __forceinline__ v8f wmma_f16(v16h a, v16h b, v8f c) {
  return __builtin_amdgcn_wmma_f32_16x16x32_f16(false, a, false, b, (short)0, c,
                                                false, false);
}

// A fragment (16x32 f16) for rows starting at row base `xrow` (row = +ln),
// K chunk at k0: halves 0..15 contiguous at k0+16*hi.
static __device__ __forceinline__ v16h load_afrag(const _Float16* xrow, int k0,
                                                  int hi) {
  v8h lo = *(const v8h*)(xrow + k0 + 16 * hi);
  v8h up = *(const v8h*)(xrow + k0 + 16 * hi + 8);
  return cat8(lo, up);
}

// ---------------------------------------------------------------- cast kernel
// 8 elements per thread (counts are multiples of 8).
__global__ void cast_f32_to_f16(const float* __restrict__ src,
                                _Float16* __restrict__ dst, int n8) {
  int i = blockIdx.x * blockDim.x + threadIdx.x;
  if (i < n8) {
    v4f a = *(const v4f*)(src + (size_t)i * 8);
    v4f b = *(const v4f*)(src + (size_t)i * 8 + 4);
    v8h o;
#pragma unroll
    for (int e = 0; e < 4; ++e) { o[e] = (_Float16)a[e]; o[4 + e] = (_Float16)b[e]; }
    *(v8h*)(dst + (size_t)i * 8) = o;
  }
}

// ------------------------------------------------------------- QKV projection
// Y = X @ W^T + bias.  X:[AM,AD] f16 row-major, W:[AD,AD] f16 row-major.
// mode 0: out[((b*H+h)*N + n)*DH + dh]   (Q, K : [B,H,N,DH])
// mode 1: out[((b*H+h)*DH + dh)*N + n]   (V transposed : [B,H,DH,N])
// One wave computes a 32x64 strip (2 row tiles x 4 col tiles). 8 waves/block.
__global__ __launch_bounds__(256)
void gemm_qkv(const _Float16* __restrict__ X, const _Float16* __restrict__ W,
              const float* __restrict__ bias, _Float16* __restrict__ out,
              int mode) {
  const int lane = threadIdx.x & 31;
  const int wave = threadIdx.x >> 5;
  const int strip = blockIdx.x * 8 + wave;          // 3072 strips total
  const int STRIPS_N = AD / 64;                     // 12
  const int m0 = (strip / STRIPS_N) * 32;
  const int n0 = (strip % STRIPS_N) * 64;
  const int hi = lane >> 4;
  const int ln = lane & 15;

  v8f z = {};
  v8f acc0[4] = {z, z, z, z};
  v8f acc1[4] = {z, z, z, z};

  const _Float16* xrow0 = X + (size_t)(m0 + ln) * AD;
  const _Float16* xrow1 = X + (size_t)(m0 + 16 + ln) * AD;
  for (int k0 = 0; k0 < AD; k0 += 32) {
    v16h a0 = load_afrag(xrow0, k0, hi);
    v16h a1 = load_afrag(xrow1, k0, hi);
#pragma unroll
    for (int t = 0; t < 4; ++t) {
      const int c = n0 + 16 * t + ln;               // output column
      // B fragment (32x16): 16 contiguous K-halves at k0+16*hi of W row c
      v16h b = *(const v16h*)(W + (size_t)c * AD + k0 + 16 * hi);
      acc0[t] = wmma_f16(a0, b, acc0[t]);
      acc1[t] = wmma_f16(a1, b, acc1[t]);
    }
  }

#pragma unroll
  for (int t = 0; t < 4; ++t) {
    const int c = n0 + 16 * t + ln;
    const float bv = bias[c];
    const int h = c / ADH, dh = c % ADH;
#pragma unroll
    for (int r = 0; r < 2; ++r) {
      const v8f* acc = r ? acc1 : acc0;
#pragma unroll
      for (int v = 0; v < 8; ++v) {
        const int m = m0 + 16 * r + v + 8 * hi;     // global row
        const int bb = m >> 10, n = m & 1023;
        const _Float16 y = (_Float16)(acc[t][v] + bv);
        size_t idx;
        if (mode == 0) idx = ((size_t)(bb * AH + h) * AN + n) * ADH + dh;
        else           idx = ((size_t)(bb * AH + h) * ADH + dh) * AN + n;
        out[idx] = y;
      }
    }
  }
}

// --------------------------------------------------------- flash attention
// One wave per (b,h, 16-row q tile); 4 waves / block. Online softmax over
// 32-key blocks; P goes through LDS to convert C-layout -> A-fragment layout.
__global__ __launch_bounds__(128)
void flash_attn(const _Float16* __restrict__ Q,   // [B,H,N,DH]
                const _Float16* __restrict__ K,   // [B,H,N,DH]
                const _Float16* __restrict__ Vt,  // [B,H,DH,N]
                _Float16* __restrict__ ctx) {     // [B,N,D]
  __shared__ __align__(64) _Float16 pbuf[4][16][32];

  const int lane = threadIdx.x & 31;
  const int wave = threadIdx.x >> 5;
  const int gw = blockIdx.x * 4 + wave;   // 6144 waves = B*H*64 exactly
  const int qt = gw & 63;
  const int bh = gw >> 6;
  const int q0 = qt * 16;
  const int hi = lane >> 4, ln = lane & 15;

  const _Float16* Qb = Q + (size_t)bh * AN * ADH;
  const _Float16* Kb = K + (size_t)bh * AN * ADH;
  const _Float16* Vb = Vt + (size_t)bh * ADH * AN;

  // Q fragments for DH=64 (two K=32 chunks), with 1/sqrt(64)=0.125 folded in.
  v16h aQ[2];
  {
    const _Float16* qrow = Qb + (size_t)(q0 + ln) * ADH;
#pragma unroll
    for (int c = 0; c < 2; ++c) {
      v16h a = load_afrag(qrow, 32 * c, hi);
      a = a * (_Float16)0.125f;   // exact power of two
      aQ[c] = a;
    }
  }

  v8f z = {};
  v8f O[4] = {z, z, z, z};
  float mrow[8], lrow[8];
#pragma unroll
  for (int v = 0; v < 8; ++v) { mrow[v] = -1e30f; lrow[v] = 0.0f; }

  for (int j0 = 0; j0 < AN; j0 += 32) {
    v8f S0 = {}, S1 = {};
#pragma unroll
    for (int c = 0; c < 2; ++c) {
      // B frag: B[k][n] = K[j0+n][32c+k]; rows of K are contiguous in dh.
      v16h b0 = *(const v16h*)(Kb + (size_t)(j0 + ln) * ADH + 32 * c + 16 * hi);
      v16h b1 = *(const v16h*)(Kb + (size_t)(j0 + 16 + ln) * ADH + 32 * c + 16 * hi);
      S0 = wmma_f16(aQ[c], b0, S0);
      S1 = wmma_f16(aQ[c], b1, S1);
    }

    // Online softmax: rows live at VGPR v (row v+8*hi), columns across 16 lanes.
    float alpha[8];
#pragma unroll
    for (int v = 0; v < 8; ++v) {
      float t = fmaxf(S0[v], S1[v]);
#pragma unroll
      for (int mask = 8; mask; mask >>= 1) t = fmaxf(t, __shfl_xor(t, mask, 32));
      const float mn = fmaxf(mrow[v], t);
      alpha[v] = __expf(mrow[v] - mn);
      mrow[v] = mn;
      const float p0 = __expf(S0[v] - mn);
      const float p1 = __expf(S1[v] - mn);
      float s = p0 + p1;
#pragma unroll
      for (int mask = 8; mask; mask >>= 1) s += __shfl_xor(s, mask, 32);
      lrow[v] = lrow[v] * alpha[v] + s;
      pbuf[wave][v + 8 * hi][ln]      = (_Float16)p0;
      pbuf[wave][v + 8 * hi][16 + ln] = (_Float16)p1;
    }
#pragma unroll
    for (int t = 0; t < 4; ++t)
#pragma unroll
      for (int v = 0; v < 8; ++v) O[t][v] *= alpha[v];

    // LDS is in-order within a wave; wait + compiler barrier before re-reading
    // the transposed P tile as an A-fragment.
    asm volatile("s_wait_dscnt 0" ::: "memory");

    v8h plo = *(const v8h*)&pbuf[wave][ln][16 * hi];
    v8h phi = *(const v8h*)&pbuf[wave][ln][16 * hi + 8];
    v16h aP = cat8(plo, phi);

#pragma unroll
    for (int t = 0; t < 4; ++t) {
      // B frag: B[k][dh] = V[j0+k][dh] = Vt[dh][j0+k]; Vt rows contiguous in n.
      v16h bV = *(const v16h*)(Vb + (size_t)(16 * t + ln) * AN + j0 + 16 * hi);
      O[t] = wmma_f16(aP, bV, O[t]);
    }
  }

  // Normalize and store ctx (f16, [B,N,D] with column h*64+dh).
  const int bb = bh / AH, h = bh % AH;
#pragma unroll
  for (int v = 0; v < 8; ++v) {
    const float inv = 1.0f / lrow[v];
    const int n = q0 + v + 8 * hi;
    _Float16* crow = ctx + ((size_t)bb * AN + n) * AD + h * ADH;
#pragma unroll
    for (int t = 0; t < 4; ++t) crow[16 * t + ln] = (_Float16)(O[t][v] * inv);
  }
}

// ---------------------------------------------------------- output projection
// d_out[m][o] = ctx[m,:] . Wo[o,:] + bo[o], f32 output. 32x64 strip per wave.
__global__ __launch_bounds__(256)
void gemm_out(const _Float16* __restrict__ X, const _Float16* __restrict__ W,
              const float* __restrict__ bias, float* __restrict__ out) {
  const int lane = threadIdx.x & 31;
  const int wave = threadIdx.x >> 5;
  const int strip = blockIdx.x * 8 + wave;
  const int STRIPS_N = AD / 64;
  const int m0 = (strip / STRIPS_N) * 32;
  const int n0 = (strip % STRIPS_N) * 64;
  const int hi = lane >> 4;
  const int ln = lane & 15;

  v8f z = {};
  v8f acc0[4] = {z, z, z, z};
  v8f acc1[4] = {z, z, z, z};

  const _Float16* xrow0 = X + (size_t)(m0 + ln) * AD;
  const _Float16* xrow1 = X + (size_t)(m0 + 16 + ln) * AD;
  for (int k0 = 0; k0 < AD; k0 += 32) {
    v16h a0 = load_afrag(xrow0, k0, hi);
    v16h a1 = load_afrag(xrow1, k0, hi);
#pragma unroll
    for (int t = 0; t < 4; ++t) {
      const int c = n0 + 16 * t + ln;
      v16h b = *(const v16h*)(W + (size_t)c * AD + k0 + 16 * hi);
      acc0[t] = wmma_f16(a0, b, acc0[t]);
      acc1[t] = wmma_f16(a1, b, acc1[t]);
    }
  }

#pragma unroll
  for (int t = 0; t < 4; ++t) {
    const int c = n0 + 16 * t + ln;
    const float bv = bias[c];
#pragma unroll
    for (int v = 0; v < 8; ++v) {
      const int mA = m0 + v + 8 * hi;
      const int mB = m0 + 16 + v + 8 * hi;
      out[(size_t)mA * AD + c] = acc0[t][v] + bv;
      out[(size_t)mB * AD + c] = acc1[t][v] + bv;
    }
  }
}

// ------------------------------------------------------------------- launcher
extern "C" void kernel_launch(void* const* d_in, const int* in_sizes, int n_in,
                              void* d_out, int out_size, void* d_ws,
                              size_t ws_size, hipStream_t stream) {
  (void)in_sizes; (void)n_in; (void)out_size; (void)ws_size;
  const float* hs = (const float*)d_in[0];
  const float* Wq = (const float*)d_in[1];
  const float* bq = (const float*)d_in[2];
  const float* Wk = (const float*)d_in[3];
  const float* bk = (const float*)d_in[4];
  const float* Wv = (const float*)d_in[5];
  const float* bv = (const float*)d_in[6];
  const float* Wo = (const float*)d_in[7];
  const float* bo = (const float*)d_in[8];

  char* ws = (char*)d_ws;
  size_t off = 0;
  auto alloc = [&](size_t bytes) {
    void* p = ws + off;
    off += (bytes + 255) & ~(size_t)255;
    return p;
  };
  _Float16* h16  = (_Float16*)alloc((size_t)AM * AD * 2);
  _Float16* wq16 = (_Float16*)alloc((size_t)AD * AD * 2);
  _Float16* wk16 = (_Float16*)alloc((size_t)AD * AD * 2);
  _Float16* wv16 = (_Float16*)alloc((size_t)AD * AD * 2);
  _Float16* wo16 = (_Float16*)alloc((size_t)AD * AD * 2);
  _Float16* q16  = (_Float16*)alloc((size_t)AM * AD * 2);
  _Float16* k16  = (_Float16*)alloc((size_t)AM * AD * 2);
  _Float16* vt16 = (_Float16*)alloc((size_t)AM * AD * 2);
  _Float16* ctx16 = h16;  // hidden f16 is dead after the QKV GEMMs

  const int nh8 = AM * AD / 8;   // 786,432
  const int nw8 = AD * AD / 8;   // 73,728
  cast_f32_to_f16<<<(nh8 + 255) / 256, 256, 0, stream>>>(hs, h16, nh8);
  cast_f32_to_f16<<<(nw8 + 255) / 256, 256, 0, stream>>>(Wq, wq16, nw8);
  cast_f32_to_f16<<<(nw8 + 255) / 256, 256, 0, stream>>>(Wk, wk16, nw8);
  cast_f32_to_f16<<<(nw8 + 255) / 256, 256, 0, stream>>>(Wv, wv16, nw8);
  cast_f32_to_f16<<<(nw8 + 255) / 256, 256, 0, stream>>>(Wo, wo16, nw8);

  // (8192/32)*(768/64) = 3072 strips, 8 waves per 256-thread block -> 384 blocks.
  gemm_qkv<<<384, 256, 0, stream>>>(h16, wq16, bq, q16, 0);
  gemm_qkv<<<384, 256, 0, stream>>>(h16, wk16, bk, k16, 0);
  gemm_qkv<<<384, 256, 0, stream>>>(h16, wv16, bv, vt16, 1);

  // B*H*(N/16) = 6144 waves, 4 waves per 128-thread block -> 1536 blocks.
  flash_attn<<<1536, 128, 0, stream>>>(q16, k16, vt16, ctx16);

  gemm_out<<<384, 256, 0, stream>>>(ctx16, wo16, bo, (float*)d_out);
}
